// MultiHeadAttention_89060441850268
// MI455X (gfx1250) — matrix operand order repaired and checked by reading, not compiled, not run on hardware
//
#include <hip/hip_runtime.h>

// MI455X (gfx1250) CDNA5 implementation: wave32 + v_wmma_f32_16x16x32_bf16
// + GLOBAL_LOAD_ASYNC_TO_LDS_B128 (ASYNCcnt) for bf16 tile staging.
// 3-kernel pipeline: QKV GEMM -> sliding-window flash attention -> out-proj GEMM.

typedef __attribute__((ext_vector_type(16))) __bf16 v16bf;
typedef __attribute__((ext_vector_type(8)))  __bf16 v8bf;
typedef __attribute__((ext_vector_type(8)))  float  v8f;

#define WMMA_BF16(a, b, c) \
  __builtin_amdgcn_wmma_f32_16x16x32_bf16(false, (a), false, (b), (short)0, (c), false, false)

// Async global->LDS copy, 16B per lane. GVS mode: saddr(u64) + vaddr(i32).
// Tracked by ASYNCcnt; drain with s_wait_asynccnt before consuming LDS.
#define ASYNC_LD_B128(ldsOff, vOff, sBase)                          \
  asm volatile("global_load_async_to_lds_b128 %0, %1, %2"           \
               :: "v"(ldsOff), "v"(vOff), "s"(sBase) : "memory")
#define WAIT_ASYNC0() asm volatile("s_wait_asynccnt 0x0" ::: "memory")
#define WAIT_DS0()    asm volatile("s_wait_dscnt 0x0" ::: "memory")

static __device__ __forceinline__ v16bf combine16(v8bf lo, v8bf hi) {
  v16bf r;
#pragma unroll
  for (int i = 0; i < 8; ++i) { r[i] = lo[i]; r[i + 8] = hi[i]; }
  return r;
}

static __device__ __forceinline__ v8f zero8() {
  v8f z;
#pragma unroll
  for (int i = 0; i < 8; ++i) z[i] = 0.0f;
  return z;
}

static __device__ __forceinline__ unsigned lds_off(const void* p) {
  // __shared__ generic pointer: low 32 bits carry the wave-relative LDS address.
  return (unsigned)(unsigned long long)p;
}

// ---------------------------------------------------------------------------
// Kernel 1: QKV projection. C[M=16384, N=3072] = x[M,1024] * qkv_w[N,1024]^T + b
// 128x128 tile per workgroup; 8 waves in 4(M) x 2(N); wave tile 32x64 (2x4 accums).
// fp32 sources must be converted to bf16 -> staged through VGPRs.
// ---------------------------------------------------------------------------
__global__ __launch_bounds__(256)
void qkv_gemm_kernel(const float* __restrict__ x, const float* __restrict__ w,
                     const float* __restrict__ bias,
                     __bf16* __restrict__ qws, __bf16* __restrict__ kws,
                     __bf16* __restrict__ vws) {
  constexpr int K = 1024, L = 4096, H = 16;
  __shared__ __bf16 lA[128 * 32];
  __shared__ __bf16 lB[128 * 32];

  const int tid = threadIdx.x;
  const int lane = tid & 31, wv = tid >> 5;
  const int lhi = lane >> 4, llo = lane & 15;
  const int m0 = blockIdx.y * 128, n0 = blockIdx.x * 128;
  const int wm = (wv >> 1) * 32, wn = (wv & 1) * 64;

  v8f acc[2][4];
#pragma unroll
  for (int a = 0; a < 2; ++a)
#pragma unroll
    for (int b2 = 0; b2 < 4; ++b2) acc[a][b2] = zero8();

  for (int kk = 0; kk < K; kk += 32) {
#pragma unroll
    for (int it = 0; it < 4; ++it) {
      int idx = tid + it * 256;
      int row = idx >> 3, c4 = idx & 7;
      float4 av = *(const float4*)(x + (size_t)(m0 + row) * K + kk + c4 * 4);
      __bf16* da = &lA[row * 32 + c4 * 4];
      da[0] = (__bf16)av.x; da[1] = (__bf16)av.y;
      da[2] = (__bf16)av.z; da[3] = (__bf16)av.w;
      float4 bv = *(const float4*)(w + (size_t)(n0 + row) * K + kk + c4 * 4);
      __bf16* db = &lB[row * 32 + c4 * 4];
      db[0] = (__bf16)bv.x; db[1] = (__bf16)bv.y;
      db[2] = (__bf16)bv.z; db[3] = (__bf16)bv.w;
    }
    __syncthreads();

    v16bf af[2], bf[4];
#pragma unroll
    for (int a = 0; a < 2; ++a) {
      int m = wm + a * 16 + llo;
      af[a] = combine16(*(const v8bf*)&lA[m * 32 + lhi * 8],
                        *(const v8bf*)&lA[m * 32 + lhi * 8 + 16]);
    }
#pragma unroll
    for (int b2 = 0; b2 < 4; ++b2) {
      int n = wn + b2 * 16 + llo;
      bf[b2] = combine16(*(const v8bf*)&lB[n * 32 + lhi * 8],
                         *(const v8bf*)&lB[n * 32 + lhi * 8 + 16]);
    }
#pragma unroll
    for (int a = 0; a < 2; ++a)
#pragma unroll
      for (int b2 = 0; b2 < 4; ++b2)
        acc[a][b2] = WMMA_BF16(af[a], bf[b2], acc[a][b2]);
    __syncthreads();
  }

#pragma unroll
  for (int a = 0; a < 2; ++a) {
#pragma unroll
    for (int b2 = 0; b2 < 4; ++b2) {
      int f = n0 + wn + b2 * 16 + llo;  // 0..3071
      float bb = bias[f];
      int h = f / 192, c = f % 192;
      int sel = c >> 6, d = c & 63;
      __bf16* dst = (sel == 0) ? qws : (sel == 1) ? kws : vws;
#pragma unroll
      for (int r = 0; r < 8; ++r) {
        int row = m0 + wm + a * 16 + r + 8 * lhi;  // flat b*L + l
        int bidx = row >> 12, lpos = row & 4095;
        float val = acc[a][b2][r] + bb;
        dst[(((size_t)bidx * H + h) * L + lpos) * 64 + d] = (__bf16)val;
      }
    }
  }
}

// ---------------------------------------------------------------------------
// Kernel 2: sliding-window flash attention. Grid (nb=32, B*H=64).
// Block = 128 queries sharing one 384-key window; 12 key tiles of 32.
// K-tile staged via async global->LDS (overlapped with V load+transpose);
// edge tiles (blocks 0/31 only) take a zero-filling fallback.
// ---------------------------------------------------------------------------
__global__ __launch_bounds__(256)
void swa_kernel(const __bf16* __restrict__ qw, const __bf16* __restrict__ kw,
                const __bf16* __restrict__ vw, const int* __restrict__ pmask,
                __bf16* __restrict__ aout) {
  constexpr int L = 4096, E = 1024;
  constexpr float NEG_INF = -1e11f, SCALE = 0.125f;  // 1/sqrt(64)
  __shared__ __bf16 ldsK[32 * 64];      // [key][dim]
  __shared__ __bf16 ldsVt[64 * 40];     // [dim][key], padded stride 40
  __shared__ __bf16 ldsP[8 * 16 * 32];  // per-wave P transpose buffer
  __shared__ int kvalid[32];

  const int tid = threadIdx.x, lane = tid & 31, wv = tid >> 5;
  const int lhi = lane >> 4, llo = lane & 15;
  const int iblk = blockIdx.x;
  const int bh = blockIdx.y;
  const int b = bh >> 4, h = bh & 15;

  // Q fragments for this wave's 16 query rows (persist across key tiles).
  const int mbase = wv * 16;
  const int qgA = iblk * 128 + mbase + llo;
  const __bf16* qrow = qw + ((size_t)bh * L + qgA) * 64;
  v16bf qf0 = combine16(*(const v8bf*)(qrow + lhi * 8),
                        *(const v8bf*)(qrow + lhi * 8 + 16));
  v16bf qf1 = combine16(*(const v8bf*)(qrow + 32 + lhi * 8),
                        *(const v8bf*)(qrow + 32 + lhi * 8 + 16));

  v8f O[4];
#pragma unroll
  for (int t = 0; t < 4; ++t) O[t] = zero8();
  float mrun[8], lrun[8];
#pragma unroll
  for (int r = 0; r < 8; ++r) { mrun[r] = -1e30f; lrun[r] = 0.0f; }

  for (int j = 0; j < 12; ++j) {
    const int kp0 = iblk * 128 - 128 + j * 32;            // uniform
    const bool fullin = (kp0 >= 0) && (kp0 + 32 <= L);    // uniform
    __syncthreads();
    if (fullin) {
      // K-tile: one async b128 per thread, tile-linear byte offset = tid*16.
      unsigned lk = lds_off(&ldsK[(tid >> 3) * 64 + (tid & 7) * 8]);
      unsigned voff = (unsigned)tid * 16u;
      ASYNC_LD_B128(lk, voff, (unsigned long long)(kw + ((size_t)bh * L + kp0) * 64));
      // V-tile through VGPRs with transposed LDS store (overlaps async K).
      int kk = tid >> 3, ch = tid & 7, d0 = ch * 8;
      uint4 vq = *(const uint4*)(vw + ((size_t)bh * L + kp0 + kk) * 64 + d0);
      __bf16 vt[8];
      *(uint4*)vt = vq;
#pragma unroll
      for (int q2 = 0; q2 < 8; ++q2) ldsVt[(d0 + q2) * 40 + kk] = vt[q2];
      if (ch == 0) kvalid[kk] = (pmask[b * L + kp0 + kk] != 0) ? 1 : 0;
      WAIT_ASYNC0();
    } else {
      // Edge tiles: zero OOB rows (avoid NaN * 0 in P*V).
      int kk = tid >> 3, ch = tid & 7, d0 = ch * 8;
      int kp = kp0 + kk;
      bool inr = (kp >= 0) && (kp < L);
      uint4 kq = make_uint4(0u, 0u, 0u, 0u), vq = make_uint4(0u, 0u, 0u, 0u);
      if (inr) {
        kq = *(const uint4*)(kw + ((size_t)bh * L + kp) * 64 + d0);
        vq = *(const uint4*)(vw + ((size_t)bh * L + kp) * 64 + d0);
      }
      *(uint4*)&ldsK[kk * 64 + d0] = kq;
      __bf16 vt[8];
      *(uint4*)vt = vq;
#pragma unroll
      for (int q2 = 0; q2 < 8; ++q2) ldsVt[(d0 + q2) * 40 + kk] = vt[q2];
      if (ch == 0) kvalid[kk] = (inr && pmask[b * L + kp] != 0) ? 1 : 0;
    }
    __syncthreads();

    // S[16q x 32k] = Q * K^T (two 16x16 accums, K-dim 64 = 2 WMMAs each)
    v8f S0 = zero8(), S1 = zero8();
    {
      int n0b = llo, n1b = 16 + llo;
      v16bf kb;
      kb = combine16(*(const v8bf*)&ldsK[n0b * 64 + lhi * 8],
                     *(const v8bf*)&ldsK[n0b * 64 + lhi * 8 + 16]);
      S0 = WMMA_BF16(qf0, kb, S0);
      kb = combine16(*(const v8bf*)&ldsK[n0b * 64 + 32 + lhi * 8],
                     *(const v8bf*)&ldsK[n0b * 64 + 32 + lhi * 8 + 16]);
      S0 = WMMA_BF16(qf1, kb, S0);
      kb = combine16(*(const v8bf*)&ldsK[n1b * 64 + lhi * 8],
                     *(const v8bf*)&ldsK[n1b * 64 + lhi * 8 + 16]);
      S1 = WMMA_BF16(qf0, kb, S1);
      kb = combine16(*(const v8bf*)&ldsK[n1b * 64 + 32 + lhi * 8],
                     *(const v8bf*)&ldsK[n1b * 64 + 32 + lhi * 8 + 16]);
      S1 = WMMA_BF16(qf1, kb, S1);
    }

    const int ok0 = kvalid[llo], ok1 = kvalid[16 + llo];
    const int kj0 = j * 32 + llo, kj1 = kj0 + 16;  // window key index 0..383

    // mask (window rel in [0,256], pad/range) + online softmax + P -> LDS
#pragma unroll
    for (int r = 0; r < 8; ++r) {
      int qi = mbase + r + 8 * lhi;  // in-block query index 0..127
      int rel0 = kj0 - qi, rel1 = kj1 - qi;
      float s0 = (ok0 && rel0 >= 0 && rel0 <= 256) ? S0[r] * SCALE : NEG_INF;
      float s1 = (ok1 && rel1 >= 0 && rel1 <= 256) ? S1[r] * SCALE : NEG_INF;
      float tmax = fmaxf(s0, s1);
#pragma unroll
      for (int off = 1; off < 16; off <<= 1)
        tmax = fmaxf(tmax, __shfl_xor(tmax, off, 32));
      float mnew = fmaxf(mrun[r], tmax);
      float corr = __expf(mrun[r] - mnew);
      float p0 = __expf(s0 - mnew);
      float p1 = __expf(s1 - mnew);
      float rs = p0 + p1;
#pragma unroll
      for (int off = 1; off < 16; off <<= 1) rs += __shfl_xor(rs, off, 32);
      lrun[r] = lrun[r] * corr + rs;
      mrun[r] = mnew;
#pragma unroll
      for (int t = 0; t < 4; ++t) O[t][r] *= corr;
      int mrow = r + 8 * lhi;
      ldsP[wv * 512 + mrow * 32 + llo] = (__bf16)p0;
      ldsP[wv * 512 + mrow * 32 + 16 + llo] = (__bf16)p1;
    }

    // In-wave LDS transpose consume: compiler fence + DS counter drain
    // (per-wave LDS ops are in-order; this blocks compiler reordering).
    WAIT_DS0();

    // P as A-fragment (16x32), V^T as B-fragments (32 keys x 16 dims)
    v16bf pf;
    {
      int mA = llo, pk = lhi * 8;
      pf = combine16(*(const v8bf*)&ldsP[wv * 512 + mA * 32 + pk],
                     *(const v8bf*)&ldsP[wv * 512 + mA * 32 + pk + 16]);
    }
#pragma unroll
    for (int t = 0; t < 4; ++t) {
      int nd = t * 16 + llo;
      v16bf vf = combine16(*(const v8bf*)&ldsVt[nd * 40 + lhi * 8],
                           *(const v8bf*)&ldsVt[nd * 40 + lhi * 8 + 16]);
      O[t] = WMMA_BF16(pf, vf, O[t]);
    }
  }

  // Epilogue: normalize, zero padded queries, write [B, L, H*64] bf16.
#pragma unroll
  for (int r = 0; r < 8; ++r) {
    int qrow2 = iblk * 128 + mbase + r + 8 * lhi;
    float inv = (lrun[r] > 0.0f) ? 1.0f / lrun[r] : 0.0f;
    if (pmask[b * L + qrow2] == 0) inv = 0.0f;
#pragma unroll
    for (int t = 0; t < 4; ++t) {
      int col = h * 64 + t * 16 + llo;
      aout[((size_t)b * L + qrow2) * E + col] = (__bf16)(O[t][r] * inv);
    }
  }
}

// ---------------------------------------------------------------------------
// Kernel 3: output projection. out[M,1024] = vals_bf16[M,1024] * o_w[1024,1024]^T + o_b
// A-tile is bf16 already -> staged with async global->LDS b128 (no VGPR trip).
// ---------------------------------------------------------------------------
__global__ __launch_bounds__(256)
void out_gemm_kernel(const __bf16* __restrict__ vals, const float* __restrict__ w,
                     const float* __restrict__ bias, float* __restrict__ out) {
  constexpr int K = 1024;
  __shared__ __bf16 lA[128 * 32];
  __shared__ __bf16 lB[128 * 32];

  const int tid = threadIdx.x;
  const int lane = tid & 31, wv = tid >> 5;
  const int lhi = lane >> 4, llo = lane & 15;
  const int m0 = blockIdx.y * 128, n0 = blockIdx.x * 128;
  const int wm = (wv >> 1) * 32, wn = (wv & 1) * 64;

  v8f acc[2][4];
#pragma unroll
  for (int a = 0; a < 2; ++a)
#pragma unroll
    for (int b2 = 0; b2 < 4; ++b2) acc[a][b2] = zero8();

  for (int kk = 0; kk < K; kk += 32) {
    // A: 512 async b128 chunks; offsets relative to tile origin (m0*K + kk).
#pragma unroll
    for (int it = 0; it < 2; ++it) {
      int idx = tid + it * 256;
      int row = idx >> 2, c8 = idx & 3;
      unsigned lk = lds_off(&lA[row * 32 + c8 * 8]);
      unsigned voff = (unsigned)(row * 2048 + c8 * 16);  // row stride K*2 bytes
      ASYNC_LD_B128(lk, voff, (unsigned long long)(vals + (size_t)m0 * K + kk));
    }
    // B: fp32 -> bf16 (must convert, stays on the VGPR path; overlaps async A)
#pragma unroll
    for (int it = 0; it < 4; ++it) {
      int idx = tid + it * 256;
      int row = idx >> 3, c4 = idx & 7;
      float4 bv = *(const float4*)(w + (size_t)(n0 + row) * K + kk + c4 * 4);
      __bf16* db = &lB[row * 32 + c4 * 4];
      db[0] = (__bf16)bv.x; db[1] = (__bf16)bv.y;
      db[2] = (__bf16)bv.z; db[3] = (__bf16)bv.w;
    }
    WAIT_ASYNC0();
    __syncthreads();

    v16bf af[2], bf[4];
#pragma unroll
    for (int a = 0; a < 2; ++a) {
      int m = wm + a * 16 + llo;
      af[a] = combine16(*(const v8bf*)&lA[m * 32 + lhi * 8],
                        *(const v8bf*)&lA[m * 32 + lhi * 8 + 16]);
    }
#pragma unroll
    for (int b2 = 0; b2 < 4; ++b2) {
      int n = wn + b2 * 16 + llo;
      bf[b2] = combine16(*(const v8bf*)&lB[n * 32 + lhi * 8],
                         *(const v8bf*)&lB[n * 32 + lhi * 8 + 16]);
    }
#pragma unroll
    for (int a = 0; a < 2; ++a)
#pragma unroll
      for (int b2 = 0; b2 < 4; ++b2)
        acc[a][b2] = WMMA_BF16(af[a], bf[b2], acc[a][b2]);
    __syncthreads();
  }

#pragma unroll
  for (int a = 0; a < 2; ++a) {
#pragma unroll
    for (int b2 = 0; b2 < 4; ++b2) {
      int f = n0 + wn + b2 * 16 + llo;
      float bb = bias[f];
#pragma unroll
      for (int r = 0; r < 8; ++r) {
        int row = m0 + wm + a * 16 + r + 8 * lhi;
        out[(size_t)row * 1024 + f] = acc[a][b2][r] + bb;
      }
    }
  }
}

// ---------------------------------------------------------------------------
extern "C" void kernel_launch(void* const* d_in, const int* in_sizes, int n_in,
                              void* d_out, int out_size, void* d_ws, size_t ws_size,
                              hipStream_t stream) {
  const float* x      = (const float*)d_in[0];  // [4,4096,1024]
  const int*   pmask  = (const int*)d_in[1];    // [4,4096]
  // d_in[2] = window_size (256, hardcoded in kernels; w=128, nb=32)
  const float* qkv_w  = (const float*)d_in[3];  // [3072,1024]
  const float* qkv_b  = (const float*)d_in[4];  // [3072]
  const float* o_w    = (const float*)d_in[5];  // [1024,1024]
  const float* o_b    = (const float*)d_in[6];  // [1024]
  float*       out    = (float*)d_out;          // [4,4096,1024]

  // Workspace: q,k,v as [B,H,L,64] bf16 + attn-out [B,L,E] bf16 = 128 MiB.
  const size_t BHLD = (size_t)4 * 16 * 4096 * 64;  // 16,777,216 elements
  __bf16* qws = (__bf16*)d_ws;
  __bf16* kws = qws + BHLD;
  __bf16* vws = kws + BHLD;
  __bf16* aws = vws + BHLD;
  (void)ws_size; (void)in_sizes; (void)n_in; (void)out_size;

  qkv_gemm_kernel<<<dim3(24, 128), 256, 0, stream>>>(x, qkv_w, qkv_b, qws, kws, vws);
  swa_kernel<<<dim3(32, 64), 256, 0, stream>>>(qws, kws, vws, pmask, aws);
  out_gemm_kernel<<<dim3(8, 128), 256, 0, stream>>>(aws, o_w, o_b, out);
}
